// ICLAttention_6356551598805
// MI455X (gfx1250) — matrix operand order repaired and checked by reading, not compiled, not run on hardware
//
#include <hip/hip_runtime.h>
#include <hip/hip_bf16.h>

// ---------------------------------------------------------------------------
// Types for CDNA5 WMMA (wave32): v16bf A/B fragments, v8f C/D fragments.
// ---------------------------------------------------------------------------
typedef __attribute__((ext_vector_type(16))) __bf16          v16bf;
typedef __attribute__((ext_vector_type(8)))  float           v8f;
typedef __attribute__((ext_vector_type(16))) unsigned short  u16x16;

#define N_HEADS 16
#define HD      64
#define S_LEN   2048
#define EMB     1024
#define BATCH   2
#define MROWS   (BATCH * S_LEN)   // 4096

// GEMM epilogue modes
#define OUT_BF16      0   // bf16 row-major [M,N]
#define OUT_F32       1   // f32 row-major [M,N]  (+bias -> final output)
#define OUT_BF16_VT   2   // bf16 scattered as vhT[b,h,d,s]  (attention V layout)

static __device__ __forceinline__ unsigned short f32_to_bf16(float f) {
    unsigned int u = __float_as_uint(f);
    unsigned int r = 0x7FFFu + ((u >> 16) & 1u);   // round-to-nearest-even
    u += r;
    return (unsigned short)(u >> 16);
}

// Build a 32-byte bf16 fragment from two 16-byte chunks.
static __device__ __forceinline__ v16bf load_frag_2x16B(const unsigned short* p0,
                                                        const unsigned short* p1) {
    u16x16 t;
    *reinterpret_cast<uint4*>(&t)       = *reinterpret_cast<const uint4*>(p0);
    *(reinterpret_cast<uint4*>(&t) + 1) = *reinterpret_cast<const uint4*>(p1);
    return __builtin_bit_cast(v16bf, t);
}

// ---------------------------------------------------------------------------
// Kernel 1: f32 -> bf16 conversion (grid-stride)
// ---------------------------------------------------------------------------
__global__ void cvt_f32_bf16_kernel(const float* __restrict__ src,
                                    unsigned short* __restrict__ dst,
                                    size_t n) {
    size_t i = (size_t)blockIdx.x * blockDim.x + threadIdx.x;
    size_t stride = (size_t)gridDim.x * blockDim.x;
    for (; i < n; i += stride) dst[i] = f32_to_bf16(src[i]);
}

// ---------------------------------------------------------------------------
// WMMA compute on one 64-wide K tile held in LDS (8 WMMAs per wave).
// ---------------------------------------------------------------------------
static __device__ __forceinline__ void compute_tile64(
        const unsigned short* __restrict__ pA,   // LDS, 64 rows x 64 K
        const unsigned short* __restrict__ pW,   // LDS, 128 rows x 64 K
        int wm, int wn, int l16, int hi, v8f acc[2][2]) {
#pragma unroll
    for (int kk = 0; kk < 64; kk += 32) {
        // A fragment (16x32) in this code's k order: half 0 holds K 0..15,
        // half 1 holds K 16..31, contiguous; row = l16 (+offsets).
        v16bf afrag[2];
#pragma unroll
        for (int i = 0; i < 2; ++i) {
            const int row  = wm + i * 16 + l16;
            const int koff = kk + (hi ? 16 : 0);
            afrag[i] = load_frag_2x16B(&pA[row * 64 + koff],
                                       &pA[row * 64 + koff + 8]);
        }
        // B fragment (32x16) in this code's k order: lane = column; half 0 K 0..15,
        // half 1 K 16..31, contiguous.
        v16bf bfrag[2];
#pragma unroll
        for (int j = 0; j < 2; ++j) {
            const int nrow = wn + j * 16 + l16;
            const int koff = kk + (hi ? 16 : 0);
            bfrag[j] = load_frag_2x16B(&pW[nrow * 64 + koff],
                                       &pW[nrow * 64 + koff + 8]);
        }
#pragma unroll
        for (int i = 0; i < 2; ++i)
#pragma unroll
            for (int j = 0; j < 2; ++j)
                acc[i][j] = __builtin_amdgcn_wmma_f32_16x16x32_bf16(
                    false, afrag[i], false, bfrag[j],
                    (short)0, acc[i][j], false, false);
    }
}

// ---------------------------------------------------------------------------
// Kernel 2: C[M,N] = A[M,K] @ W[N,K]^T (+bias), bf16 in, f32 accumulate.
// Block: 256 thr (8 waves). Block tile 64x128, wave tile 32x32 (2x2 WMMA).
// Software pipeline: global->regs for tile k+1 overlapped with WMMA on LDS
// tile k; ping-pong LDS buffers (48KB); K-step 64; one barrier/iteration.
// Staging uses explicit scalar uint4 registers (no arrays/lambdas) and the
// loop is peeled so the steady state has no conditionals -> no scratch.
// ---------------------------------------------------------------------------
__global__ __launch_bounds__(256, 1)
void gemm_bf16_kernel(const unsigned short* __restrict__ A,   // [M,K] bf16
                      const unsigned short* __restrict__ W,   // [N,K] bf16
                      const float* __restrict__ bias,         // [N] or null
                      void* __restrict__ Cout,
                      int M, int N, int K, int out_mode) {
    __shared__ __align__(16) unsigned short sA[2][64 * 64];    // 2 x 8KB
    __shared__ __align__(16) unsigned short sW[2][128 * 64];   // 2 x 16KB

    const int tid  = threadIdx.x;
    const int lane = tid & 31;
    const int wid  = tid >> 5;
    const int l16  = lane & 15;
    const int hi   = lane >> 4;

    const int bm = blockIdx.x % (M / 64);
    const int bn = blockIdx.x / (M / 64);
    const int m0 = bm * 64;
    const int n0 = bn * 128;
    const int wm = (wid >> 2) * 32;
    const int wn = (wid & 3) * 32;

    // Per-thread staging assignments (K-step 64):
    //   A tile 64x64: 16 elems/thread (2 x b128)
    //   W tile 128x64: 32 elems/thread (4 x b128)
    const int aRow = (tid * 16) >> 6;
    const int aCol = (tid * 16) & 63;
    const int wRow = tid >> 1;
    const int wCol = (tid & 1) * 32;

    const unsigned short* gA = &A[(size_t)(m0 + aRow) * K + aCol];
    const unsigned short* gW = &W[(size_t)(n0 + wRow) * K + wCol];
    unsigned short* lA = &sA[0][aRow * 64 + aCol];
    unsigned short* lW = &sW[0][wRow * 64 + wCol];

    v8f acc[2][2] = {};
    uint4 ra0, ra1, rw0, rw1, rw2, rw3;

    // ---- prologue: tile 0 -> regs -> LDS buf 0 ----
    {
        const uint4* s = reinterpret_cast<const uint4*>(gA);
        ra0 = s[0]; ra1 = s[1];
        const uint4* t = reinterpret_cast<const uint4*>(gW);
        rw0 = t[0]; rw1 = t[1]; rw2 = t[2]; rw3 = t[3];
        uint4* da = reinterpret_cast<uint4*>(lA);
        da[0] = ra0; da[1] = ra1;
        uint4* dw = reinterpret_cast<uint4*>(lW);
        dw[0] = rw0; dw[1] = rw1; dw[2] = rw2; dw[3] = rw3;
    }
    __syncthreads();

    const int nk = K >> 6;
    for (int kt = 0; kt < nk - 1; ++kt) {
        const int cur = kt & 1;
        // issue global loads for tile kt+1 (latency hidden by WMMAs below)
        const uint4* s = reinterpret_cast<const uint4*>(gA + (size_t)(kt + 1) * 64);
        ra0 = s[0]; ra1 = s[1];
        const uint4* t = reinterpret_cast<const uint4*>(gW + (size_t)(kt + 1) * 64);
        rw0 = t[0]; rw1 = t[1]; rw2 = t[2]; rw3 = t[3];
        __builtin_prefetch(s + 8, 0, 3);
        __builtin_prefetch(t + 8, 0, 3);

        compute_tile64(sA[cur], sW[cur], wm, wn, l16, hi, acc);

        uint4* da = reinterpret_cast<uint4*>(lA + (cur ^ 1) * (64 * 64));
        da[0] = ra0; da[1] = ra1;
        uint4* dw = reinterpret_cast<uint4*>(lW + (cur ^ 1) * (128 * 64));
        dw[0] = rw0; dw[1] = rw1; dw[2] = rw2; dw[3] = rw3;
        __syncthreads();
    }
    compute_tile64(sA[(nk - 1) & 1], sW[(nk - 1) & 1], wm, wn, l16, hi, acc);

    // ---- epilogue: C layout VGPR r = row r (lanes 0-15) / r+8 (hi lanes) ----
#pragma unroll
    for (int i = 0; i < 2; ++i)
#pragma unroll
        for (int j = 0; j < 2; ++j)
#pragma unroll
            for (int r = 0; r < 8; ++r) {
                const int m = m0 + wm + i * 16 + r + hi * 8;
                const int n = n0 + wn + j * 16 + l16;
                float v = acc[i][j][r];
                if (bias) v += bias[n];
                if (out_mode == OUT_F32) {
                    reinterpret_cast<float*>(Cout)[(size_t)m * N + n] = v;
                } else if (out_mode == OUT_BF16) {
                    reinterpret_cast<unsigned short*>(Cout)[(size_t)m * N + n] =
                        f32_to_bf16(v);
                } else {  // OUT_BF16_VT: vhT[b, h, d, s], s contiguous
                    const int bb = m >> 11;       // m / S_LEN
                    const int ss = m & (S_LEN - 1);
                    const int hh = n >> 6;        // n / HD
                    const int dd = n & (HD - 1);
                    reinterpret_cast<unsigned short*>(Cout)[
                        (((size_t)bb * N_HEADS + hh) * HD + dd) * S_LEN + ss] =
                        f32_to_bf16(v);
                }
            }
}

// ---------------------------------------------------------------------------
// Kernel 3: flash attention. Each wave owns 16 query rows of one (b,h).
// Scores via q@k^T WMMA, online softmax with width-16 shuffles, P restaged
// through a per-wave LDS tile into A-fragment layout for P@V WMMAs.
// qh/kh/out layout: [B*S, 1024], head h at cols h*64..h*64+63.
// vhT layout: [B, H, HD, S] (t contiguous) -> V B-fragments are two b128s.
// ---------------------------------------------------------------------------
__global__ __launch_bounds__(256, 1)
void flash_attn_kernel(const unsigned short* __restrict__ qh,
                       const unsigned short* __restrict__ kh,
                       const unsigned short* __restrict__ vhT,
                       unsigned short* __restrict__ out) {
    __shared__ __align__(16) unsigned short sP[8][16 * 32];  // per-wave P tile

    const int tid  = threadIdx.x;
    const int lane = tid & 31;
    const int wid  = tid >> 5;
    const int l16  = lane & 15;
    const int hi   = lane >> 4;

    const int nq  = S_LEN / 128;
    const int b   = blockIdx.x / (N_HEADS * nq);
    const int rem = blockIdx.x % (N_HEADS * nq);
    const int h   = rem / nq;
    const int qc  = rem % nq;
    const int r0  = qc * 128 + wid * 16;         // wave's query row base

    const size_t rs = EMB;
    const unsigned short* qb  = qh  + (size_t)b * S_LEN * rs + h * HD;
    const unsigned short* kb  = kh  + (size_t)b * S_LEN * rs + h * HD;
    const unsigned short* vtb = vhT + ((size_t)b * N_HEADS + h) * HD * S_LEN;

    // q A-fragments for d-halves 0..31 / 32..63 (resident)
    v16bf aq[2];
#pragma unroll
    for (int dh = 0; dh < 2; ++dh) {
        const unsigned short* p = qb + (size_t)(r0 + l16) * rs + dh * 32 + hi * 16;
        aq[dh] = load_frag_2x16B(p, p + 8);
    }

    v8f o[4] = {};                 // 16x64 output accumulator (4 d-tiles)
    float mrow[8], lrow[8], sc[8];
#pragma unroll
    for (int r = 0; r < 8; ++r) { mrow[r] = -1e30f; lrow[r] = 0.0f; }

    const int tmax = r0 + 14;      // largest attended key index for this wave
    for (int tb = 0; tb <= tmax; tb += 32) {
        // ---- scores S = (q @ k^T), two 16x16 col-halves ----
        v8f s[2] = {};
#pragma unroll
        for (int j = 0; j < 2; ++j)
#pragma unroll
            for (int dh = 0; dh < 2; ++dh) {
                const int t = tb + j * 16 + l16;
                const unsigned short* p = kb + (size_t)t * rs + dh * 32 + hi * 16;
                v16bf bk = load_frag_2x16B(p, p + 8);
                s[j] = __builtin_amdgcn_wmma_f32_16x16x32_bf16(
                    false, aq[dh], false, bk, (short)0, s[j], false, false);
            }
        // ---- causal mask + scale (1/sqrt(64)); [0,0] stays unmasked ----
#pragma unroll
        for (int j = 0; j < 2; ++j)
#pragma unroll
            for (int r = 0; r < 8; ++r) {
                const int m = r0 + r + hi * 8;
                const int t = tb + j * 16 + l16;
                const float v = s[j][r] * 0.125f;
                const bool masked = (t >= m) && !((m == 0) && (t == 0));
                s[j][r] = masked ? -1e30f : v;
            }
        // ---- online softmax row stats (row = 16 lanes of one half) ----
#pragma unroll
        for (int r = 0; r < 8; ++r) {
            float mx = fmaxf(s[0][r], s[1][r]);
            for (int off = 1; off < 16; off <<= 1)
                mx = fmaxf(mx, __shfl_xor(mx, off, 16));
            const float mn = fmaxf(mrow[r], mx);
            sc[r] = __expf(mrow[r] - mn);
            mrow[r] = mn;
            const float p0 = __expf(s[0][r] - mn);
            const float p1 = __expf(s[1][r] - mn);
            s[0][r] = p0; s[1][r] = p1;
            float sum = p0 + p1;
            for (int off = 1; off < 16; off <<= 1)
                sum += __shfl_xor(sum, off, 16);
            lrow[r] = lrow[r] * sc[r] + sum;
        }
        // ---- rescale O ----
#pragma unroll
        for (int dt = 0; dt < 4; ++dt)
#pragma unroll
            for (int r = 0; r < 8; ++r) o[dt][r] *= sc[r];

        // ---- P: C-layout -> LDS (row,col) -> A-fragment ----
        unsigned short* Pw = sP[wid];
#pragma unroll
        for (int j = 0; j < 2; ++j)
#pragma unroll
            for (int r = 0; r < 8; ++r)
                Pw[(r + hi * 8) * 32 + j * 16 + l16] = f32_to_bf16(s[j][r]);
        __builtin_amdgcn_wave_barrier();   // keep DS store->load order
        const unsigned short* pp = &Pw[l16 * 32 + hi * 16];
        v16bf ap = load_frag_2x16B(pp, pp + 8);

        // ---- O += P @ V : B frag = row of vhT (t contiguous), two b128s ----
#pragma unroll
        for (int dt = 0; dt < 4; ++dt) {
            const int d = dt * 16 + l16;
            const unsigned short* vp = vtb + (size_t)d * S_LEN + tb + hi * 16;
            v16bf bvf = load_frag_2x16B(vp, vp + 8);
            o[dt] = __builtin_amdgcn_wmma_f32_16x16x32_bf16(
                false, ap, false, bvf, (short)0, o[dt], false, false);
        }
    }

    // ---- normalize and store bf16 (layout [b,s, h*64+d]) ----
#pragma unroll
    for (int r = 0; r < 8; ++r) lrow[r] = 1.0f / lrow[r];
    unsigned short* ob = out + (size_t)b * S_LEN * rs + h * HD;
#pragma unroll
    for (int dt = 0; dt < 4; ++dt)
#pragma unroll
        for (int r = 0; r < 8; ++r)
            ob[(size_t)(r0 + r + hi * 8) * rs + dt * 16 + l16] =
                f32_to_bf16(o[dt][r] * lrow[r]);
}

// ---------------------------------------------------------------------------
// Host-side orchestration
// ---------------------------------------------------------------------------
extern "C" void kernel_launch(void* const* d_in, const int* in_sizes, int n_in,
                              void* d_out, int out_size, void* d_ws, size_t ws_size,
                              hipStream_t stream) {
    const float* q  = (const float*)d_in[0];
    const float* k  = (const float*)d_in[1];
    const float* v  = (const float*)d_in[2];
    const float* Wq = (const float*)d_in[3];
    const float* Wk = (const float*)d_in[4];
    const float* Wv = (const float*)d_in[5];
    const float* bv = (const float*)d_in[6];
    const float* Wo = (const float*)d_in[7];
    const float* bo = (const float*)d_in[8];
    float* outp = (float*)d_out;

    const size_t MK = (size_t)MROWS * EMB;   // 4096*1024
    const size_t NK = (size_t)EMB * EMB;     // 1024*1024

    unsigned short* w    = (unsigned short*)d_ws;
    unsigned short* qb   = w;
    unsigned short* kb   = qb + MK;
    unsigned short* vbuf = kb + MK;
    unsigned short* Wqb  = vbuf + MK;
    unsigned short* Wkb  = Wqb + NK;
    unsigned short* Wvb  = Wkb + NK;
    unsigned short* Wob  = Wvb + NK;
    unsigned short* qhh  = Wob + NK;
    unsigned short* khh  = qhh + MK;
    unsigned short* vhT  = khh + MK;   // [B, H, HD, S]
    unsigned short* att  = vhT + MK;

    // 1) convert everything to bf16 once
    cvt_f32_bf16_kernel<<<1024, 256, 0, stream>>>(q,  qb,  MK);
    cvt_f32_bf16_kernel<<<1024, 256, 0, stream>>>(k,  kb,  MK);
    cvt_f32_bf16_kernel<<<1024, 256, 0, stream>>>(v,  vbuf,MK);
    cvt_f32_bf16_kernel<<<1024, 256, 0, stream>>>(Wq, Wqb, NK);
    cvt_f32_bf16_kernel<<<1024, 256, 0, stream>>>(Wk, Wkb, NK);
    cvt_f32_bf16_kernel<<<1024, 256, 0, stream>>>(Wv, Wvb, NK);
    cvt_f32_bf16_kernel<<<1024, 256, 0, stream>>>(Wo, Wob, NK);

    // 2) projections: qh/kh row-major, V written pre-transposed for attention
    const int gemm_grid = (MROWS / 64) * (EMB / 128);  // 512
    gemm_bf16_kernel<<<gemm_grid, 256, 0, stream>>>(qb,  Wqb, nullptr, qhh,
                                                    MROWS, EMB, EMB, OUT_BF16);
    gemm_bf16_kernel<<<gemm_grid, 256, 0, stream>>>(kb,  Wkb, nullptr, khh,
                                                    MROWS, EMB, EMB, OUT_BF16);
    gemm_bf16_kernel<<<gemm_grid, 256, 0, stream>>>(vbuf,Wvb, bv,      vhT,
                                                    MROWS, EMB, EMB, OUT_BF16_VT);

    // 3) flash attention: 512 blocks = B * H * (S/128)
    flash_attn_kernel<<<BATCH * N_HEADS * (S_LEN / 128), 256, 0, stream>>>(
        qhh, khh, vhT, att);

    // 4) output projection -> f32 d_out
    gemm_bf16_kernel<<<gemm_grid, 256, 0, stream>>>(att, Wob, bo, outp,
                                                    MROWS, EMB, EMB, OUT_F32);
}
